// MixedOperation_49314814492681
// MI455X (gfx1250) — matrix operand order repaired and checked by reading, not compiled, not run on hardware
//
#include <hip/hip_runtime.h>
#include <hip/hip_bf16.h>

typedef __attribute__((ext_vector_type(16))) _Float16 v16h;
typedef __attribute__((ext_vector_type(8)))  _Float16 v8h;
typedef __attribute__((ext_vector_type(8)))  float    v8f;

#define N_OPS 8
#define CCH   128          // channels
#define HW    1024         // 32*32
#define OUT_ELEMS (16*128*32*32)

// ---------------- workspace layout (bytes) ----------------
// [0,32)           : m[8] f32
// [128,640)        : b_eff[128] f32
// [1024, +1605632) : W_eff f16  [o][tap(49)][ci(128)]
// [1606656, +4194304) : x_f16   [b][y][x][ci]   channels-last
#define WS_M_OFF     0
#define WS_BEFF_OFF  128
#define WS_WT_OFF    1024
#define WS_XH_OFF    1606656

// ============================================================
// Phase 1: gumbel-softmax m, scalar accumulators, effective bias
// ============================================================
__global__ void prep_kernel(const float* __restrict__ thetas,
                            const float* __restrict__ gum,
                            const float* __restrict__ tau,
                            const float* __restrict__ f0,
                            const float* __restrict__ p0,
                            const float* __restrict__ fc,
                            const float* __restrict__ pc,
                            const float* __restrict__ bias,   // [8][128]
                            float* __restrict__ out,
                            float* __restrict__ ws_m,
                            float* __restrict__ ws_beff) {
  __shared__ float sm[N_OPS];
  const int tid = threadIdx.x;
  if (tid == 0) {
    float th[N_OPS], lg[N_OPS], m[N_OPS];
    float mx = -1e30f;
    for (int i = 0; i < N_OPS; ++i) { th[i] = thetas[i]; mx = fmaxf(mx, th[i]); }
    float se = 0.f;
    for (int i = 0; i < N_OPS; ++i) se += expf(th[i] - mx);
    const float lse = mx + logf(se);
    const float t = tau[0];
    float mx2 = -1e30f;
    for (int i = 0; i < N_OPS; ++i) {
      lg[i] = (th[i] - lse + gum[i]) / t;
      mx2 = fmaxf(mx2, lg[i]);
    }
    float se2 = 0.f;
    for (int i = 0; i < N_OPS; ++i) { m[i] = expf(lg[i] - mx2); se2 += m[i]; }
    float fa = f0[0], pa = p0[0];
    for (int i = 0; i < N_OPS; ++i) {
      m[i] /= se2;
      sm[i] = m[i];
      ws_m[i] = m[i];
      fa += m[i] * fc[i];
      pa += m[i] * pc[i];
    }
    out[OUT_ELEMS + 0] = fa;
    out[OUT_ELEMS + 1] = pa;
  }
  __syncthreads();
  if (tid < CCH) {
    float be = 0.f;
    #pragma unroll
    for (int i = 0; i < N_OPS; ++i) be += sm[i] * bias[i * CCH + tid];
    ws_beff[tid] = be;
  }
}

// ============================================================
// Phase 2: combine 8 kernels into one effective 7x7 (f16)
// W_eff layout: [o][tap=r*7+s][ci]
// ============================================================
__global__ void wcombine_kernel(const float* __restrict__ w1,
                                const float* __restrict__ w3,
                                const float* __restrict__ w5,
                                const float* __restrict__ w7,
                                const float* __restrict__ ws_m,
                                _Float16* __restrict__ wt) {
  __shared__ float sm[N_OPS];
  if (threadIdx.x < N_OPS) sm[threadIdx.x] = ws_m[threadIdx.x];
  __syncthreads();
  const int e = blockIdx.x * 256 + threadIdx.x;   // 128*49*128 = 802816 exact
  const int ci  = e & 127;
  const int tap = (e >> 7) % 49;
  const int o   = e / (49 * 128);
  const int r = tap / 7, s = tap % 7;
  const int oc = o * CCH + ci;

  float acc = sm[6] * w7[oc * 49 + r * 7 + s]
            + sm[7] * w7[128 * 128 * 49 + oc * 49 + r * 7 + s];
  if (r >= 1 && r <= 5 && s >= 1 && s <= 5) {
    const int rr = r - 1, ss = s - 1;
    acc += sm[4] * w5[oc * 25 + rr * 5 + ss]
         + sm[5] * w5[128 * 128 * 25 + oc * 25 + rr * 5 + ss];
  }
  if (r >= 2 && r <= 4 && s >= 2 && s <= 4) {
    const int rr = r - 2, ss = s - 2;
    acc += sm[2] * w3[oc * 9 + rr * 3 + ss]
         + sm[3] * w3[128 * 128 * 9 + oc * 9 + rr * 3 + ss];
  }
  if (r == 3 && s == 3) {
    acc += sm[0] * w1[oc] + sm[1] * w1[128 * 128 + oc];
  }
  wt[(o * 49 + tap) * CCH + ci] = (_Float16)acc;
}

// ============================================================
// Phase 3: x f32 NCHW -> f16 channels-last [b][y][x][ci]
// 32(ch) x 32(yx) LDS tile: coalesced f32 reads, 16B vector writes.
// grid: 16 b * 4 cgroup * 32 yxgroup = 2048 blocks, 256 thr
// ============================================================
__global__ void xconv_kernel(const float* __restrict__ x,
                             _Float16* __restrict__ xh) {
  __shared__ _Float16 lt[32 * 40];     // [yx][c], stride 40 halves
  const int tid = threadIdx.x;
  const int blk = blockIdx.x;
  const int b   = blk >> 7;
  const int c0  = ((blk >> 5) & 3) << 5;
  const int yx0 = (blk & 31) << 5;
  #pragma unroll
  for (int i = 0; i < 4; ++i) {
    const int u  = tid + i * 256;
    const int c  = u >> 5;
    const int yx = u & 31;
    lt[yx * 40 + c] = (_Float16)x[(size_t)(b * CCH + c0 + c) * HW + yx0 + yx];
  }
  __syncthreads();
  if (tid < 128) {
    const int yx = tid >> 2;
    const int cg = tid & 3;
    const v8h v = *(const v8h*)(&lt[yx * 40 + cg * 8]);
    *(v8h*)(xh + (size_t)(b * HW + yx0 + yx) * CCH + c0 + cg * 8) = v;
  }
}

// ============================================================
// Phase 4: implicit-GEMM 7x7 conv via v_wmma_f32_16x16x32_f16
// block = 128 thr (4 waves) covers 2 rows x 32 cols of one image.
// K chunked by 32 channels (4 phases); per tap one WMMA-K step.
// Weights for the current tap staged in LDS (shared by all 4 waves),
// double-buffered + register-prefetched: 1 barrier per tap.
// ============================================================
#define CH_STRIDE 40            // halves per (row,col) slot: 80B (16B aligned)
#define LDS_COLS  40            // 38 used (x in -3..34)
#define LDS_ROWS  8             // y0-3 .. y0+4
#define B_STRIDE  40            // halves per output channel in weight stage

__global__ __launch_bounds__(128)
void conv_main_kernel(const _Float16* __restrict__ xh,
                      const _Float16* __restrict__ wt,
                      const float* __restrict__ beff,
                      float* __restrict__ out) {
  __shared__ _Float16 halo[LDS_ROWS * LDS_COLS * CH_STRIDE]; // 25,600 B
  __shared__ _Float16 bst[2][CCH * B_STRIDE];                // 2 x 10,240 B

  const int tid  = threadIdx.x;
  const int blk  = blockIdx.x;        // 0..255
  const int bimg = blk >> 4;
  const int y0   = (blk & 15) << 1;
  const int w    = tid >> 5;          // wave 0..3
  const int lane = tid & 31;
  const int M    = lane & 15;         // A row (pixel) == C column (N)
  const int hi   = lane >> 4;
  const int rowofs = w >> 1;          // 0/1
  const int xbase  = (w & 1) << 4;    // 0/16
  const int pixrow_base = bimg * HW;

  v8f acc[8];
  #pragma unroll
  for (int n = 0; n < 8; ++n) acc[n] = (v8f){0.f,0.f,0.f,0.f,0.f,0.f,0.f,0.f};

  for (int h4 = 0; h4 < 4; ++h4) {            // 32-channel K chunks
    __syncthreads();                           // prior readers of halo/bst done
    // ---- stage halo: 8 rows x 38 cols x 32 ch (b128 chunks) ----
    for (int u = tid; u < LDS_ROWS * 38 * 4; u += 128) {
      const int ch8 = u & 3;
      const int col = (u >> 2) % 38;
      const int row = (u >> 2) / 38;
      const int sy = y0 - 3 + row;
      const int sx = col - 3;
      v8h val = (v8h){0,0,0,0,0,0,0,0};
      if (sy >= 0 && sy < 32 && sx >= 0 && sx < 32) {
        val = *(const v8h*)(xh + ((size_t)(pixrow_base + sy * 32 + sx) * CCH
                                  + h4 * 32 + ch8 * 8));
      }
      *(v8h*)(&halo[(row * LDS_COLS + col) * CH_STRIDE + ch8 * 8]) = val;
    }
    // ---- prologue: weights for tap 0 into buffer 0 (thread <-> o) ----
    {
      v8h pre[4];
      #pragma unroll
      for (int i = 0; i < 4; ++i)
        pre[i] = *(const v8h*)(wt + ((size_t)(tid * 49) * CCH + h4 * 32 + i * 8));
      #pragma unroll
      for (int i = 0; i < 4; ++i)
        *(v8h*)(&bst[0][tid * B_STRIDE + i * 8]) = pre[i];
    }
    __syncthreads();                           // halo + buf0 visible

    for (int tap = 0; tap < 49; ++tap) {
      const int cur = tap & 1;
      const int nxt = cur ^ 1;
      // prefetch next tap's weights into registers (hides L2 latency)
      v8h pre[4];
      if (tap < 48) {
        #pragma unroll
        for (int i = 0; i < 4; ++i)
          pre[i] = *(const v8h*)(wt + ((size_t)(tid * 49 + tap + 1) * CCH
                                       + h4 * 32 + i * 8));
      }
      // A fragment: 16-bit A layout, lane half hi: K = hi*8+[0..7], 16+hi*8+[0..7]
      const int r = tap / 7;
      const int s = tap - r * 7;
      const int ly = rowofs + r;               // 0..7
      const int lx = xbase + M + s;            // 0..37
      const int pixoff = (ly * LDS_COLS + lx) * CH_STRIDE;
      const v8h alo = *(const v8h*)(&halo[pixoff + hi * 8]);
      const v8h ahi = *(const v8h*)(&halo[pixoff + 16 + hi * 8]);
      v16h a;
      #pragma unroll
      for (int j = 0; j < 8; ++j) { a[j] = alo[j]; a[j + 8] = ahi[j]; }

      #pragma unroll
      for (int n = 0; n < 8; ++n) {
        const int o = (n << 4) + M;            // here M acts as N column
        const v8h blo = *(const v8h*)(&bst[cur][o * B_STRIDE + hi * 8]);
        const v8h bhi = *(const v8h*)(&bst[cur][o * B_STRIDE + 16 + hi * 8]);
        v16h bf;
        #pragma unroll
        for (int j = 0; j < 8; ++j) { bf[j] = blo[j]; bf[j + 8] = bhi[j]; }
        acc[n] = __builtin_amdgcn_wmma_f32_16x16x32_f16(
            false, a, false, bf, (short)0, acc[n], false, false);
      }
      // back-buffer store: its last readers (tap-1) passed the previous barrier
      if (tap < 48) {
        #pragma unroll
        for (int i = 0; i < 4; ++i)
          *(v8h*)(&bst[nxt][tid * B_STRIDE + i * 8]) = pre[i];
      }
      __syncthreads();                         // make buf[nxt] visible for tap+1
    }
  }

  // ---- epilogue: C layout lane N = lane&15, VGPR v -> pixel v + 8*hi ----
  const int y = y0 + rowofs;
  #pragma unroll
  for (int n = 0; n < 8; ++n) {
    const int o = (n << 4) + M;
    const float bv = beff[o];
    #pragma unroll
    for (int v = 0; v < 8; ++v) {
      const int xcol = xbase + v + hi * 8;
      out[((size_t)(bimg * CCH + o) * 32 + y) * 32 + xcol] = acc[n][v] + bv;
    }
  }
}

// ============================================================
extern "C" void kernel_launch(void* const* d_in, const int* in_sizes, int n_in,
                              void* d_out, int out_size, void* d_ws, size_t ws_size,
                              hipStream_t stream) {
  const float* x      = (const float*)d_in[0];
  const float* tau    = (const float*)d_in[1];
  const float* f0     = (const float*)d_in[2];
  const float* p0     = (const float*)d_in[3];
  const float* thetas = (const float*)d_in[4];
  const float* gum    = (const float*)d_in[5];
  const float* fc     = (const float*)d_in[6];
  const float* pc     = (const float*)d_in[7];
  const float* w1     = (const float*)d_in[8];
  const float* w3     = (const float*)d_in[9];
  const float* w5     = (const float*)d_in[10];
  const float* w7     = (const float*)d_in[11];
  const float* bias   = (const float*)d_in[12];

  float* out = (float*)d_out;
  char*  ws  = (char*)d_ws;
  float*    ws_m    = (float*)(ws + WS_M_OFF);
  float*    ws_beff = (float*)(ws + WS_BEFF_OFF);
  _Float16* wt      = (_Float16*)(ws + WS_WT_OFF);
  _Float16* xh      = (_Float16*)(ws + WS_XH_OFF);

  prep_kernel<<<1, 128, 0, stream>>>(thetas, gum, tau, f0, p0, fc, pc, bias,
                                     out, ws_m, ws_beff);
  wcombine_kernel<<<(128 * 49 * 128) / 256, 256, 0, stream>>>(w1, w3, w5, w7,
                                                              ws_m, wt);
  xconv_kernel<<<2048, 256, 0, stream>>>(x, xh);
  conv_main_kernel<<<256, 128, 0, stream>>>(xh, wt, ws_beff, out);
}
